// SlotAttention_46832323395949
// MI455X (gfx1250) — matrix operand order repaired and checked by reading, not compiled, not run on hardware
//
#include <hip/hip_runtime.h>
#include <hip/hip_bf16.h>

// ---------------------------------------------------------------------------
// Slot attention on MI455X (gfx1250, wave32).
// Big GEMMs (N=4096 rows x 64 x 64) run on v_wmma_f32_16x16x32_f16 with
// A/B tiles staged row-major in LDS, fragments read in contiguous k order.
// LDS rows padded to 160B so all fragment accesses are aligned b128 vectors.
// All reductions are deterministic (no FP atomics).
// ---------------------------------------------------------------------------

typedef __attribute__((ext_vector_type(16))) _Float16 v16h;
typedef __attribute__((ext_vector_type(8)))  _Float16 v8h;
typedef __attribute__((ext_vector_type(8)))  float    v8f;

#define NB 8          // batch
#define NPIX 4096     // H*W
#define DD 64         // SLOT_DIM == IN_DIM
#define NK 8          // slots (1 bg + 7 fg)
#define SCALE 0.125f  // 64^-0.5
#define EPS_ATTN 1e-8f
#define LN_EPS 1e-5f
#define LDSP 80       // padded LDS row (halves): 160B = 32B-aligned rows

// ---- workspace offsets (floats) ----
constexpr size_t OF_FEATN  = 0;                                   // 8*4096*64
constexpr size_t OF_COLORN = OF_FEATN  + (size_t)NB*NPIX*DD;      // 8*4096*8
constexpr size_t OF_KFGB   = OF_COLORN + (size_t)NB*NPIX*8;
constexpr size_t OF_VFGB   = OF_KFGB   + (size_t)NB*NPIX*DD;
constexpr size_t OF_KBT    = OF_VFGB   + (size_t)NB*NPIX*DD;
constexpr size_t OF_VBT    = OF_KBT    + (size_t)NB*NPIX*DD;
constexpr size_t OF_KBG    = OF_VBT    + (size_t)NB*NPIX*DD;
constexpr size_t OF_VBG    = OF_KBG    + (size_t)NB*NPIX*DD;
constexpr size_t OF_LOGITS = OF_VBG    + (size_t)NB*NPIX*DD;      // B*8*N
constexpr size_t OF_SUMS   = OF_LOGITS + (size_t)NB*NK*NPIX;      // 64
constexpr size_t OF_POSA   = OF_SUMS + 64;                        // 112 (pad 128)
constexpr size_t OF_POSB   = OF_POSA + 128;
constexpr size_t OF_QFG    = OF_POSB + 128;                       // 8*7*64
constexpr size_t OF_QBG    = OF_QFG + 3584;                       // 8*64
constexpr size_t OF_SFG    = OF_QBG + 512;                        // slots fg
constexpr size_t OF_SBG    = OF_SFG + 3584;                       // slots bg
constexpr size_t OF_UFG    = OF_SBG + 512;                        // upd fg
constexpr size_t OF_UBG    = OF_UFG + 3584;                       // upd bg
constexpr size_t OF_UPART  = OF_UBG + 512;                        // 56*64*64 partials

// ---- fragment helpers (this code's contiguous k order for 16x16x32 f16, wave32) ----
// A (16x32, MxK): lane L holds row M=L&15; halves e -> K = (L<16?0:16) + e
// B (32x16, KxN): lane L holds col N=L&15; halves e -> K = (L<16?0:16) + e
// C/D (16x16 f32): VGPR r <-> M = r + (L<16?0:8); N = L&15

__device__ __forceinline__ v16h load_a_frag(const _Float16* rowp, int ka) {
  v8h a0 = *(const v8h*)(rowp + ka);
  v8h a1 = *(const v8h*)(rowp + ka + 8);
  return __builtin_shufflevector(a0, a1, 0, 1, 2, 3, 4, 5, 6, 7,
                                 8, 9, 10, 11, 12, 13, 14, 15);
}

// ===========================================================================
// init: slots = mu + exp(logsigma)*noise ; pos = broadcast(fg_position)
// ===========================================================================
__global__ void init_kernel(const float* __restrict__ mu, const float* __restrict__ ls,
                            const float* __restrict__ mubg, const float* __restrict__ lsbg,
                            const float* __restrict__ nfg, const float* __restrict__ nbg,
                            const float* __restrict__ pos0,
                            float* __restrict__ sfg, float* __restrict__ sbg,
                            float* __restrict__ pos) {
  int i = blockIdx.x * 256 + threadIdx.x;
  if (i < NB * 7 * DD) sfg[i] = mu[i & 63] + __expf(ls[i & 63]) * nfg[i];
  if (i < NB * DD)     sbg[i] = mubg[i & 63] + __expf(lsbg[i & 63]) * nbg[i];
  if (i < NB * 7 * 2)  pos[i] = pos0[i % 14];
}

// ===========================================================================
// per-row layernorm, 64 features (feat) / 8 features (color)
// ===========================================================================
__global__ void ln_feat_kernel(const float* __restrict__ x, const float* __restrict__ g,
                               const float* __restrict__ b, float* __restrict__ y) {
  int row = blockIdx.x * 256 + threadIdx.x;
  if (row >= NB * NPIX) return;
  const float* xp = x + (size_t)row * DD;
  float s = 0.f, ss = 0.f;
  for (int i = 0; i < DD; i++) { float v = xp[i]; s += v; ss += v * v; }
  float m = s * (1.f / DD);
  float rstd = rsqrtf(ss * (1.f / DD) - m * m + LN_EPS);
  float* yp = y + (size_t)row * DD;
  for (int i = 0; i < DD; i++) yp[i] = (xp[i] - m) * rstd * g[i] + b[i];
}

__global__ void ln_color_kernel(const float* __restrict__ x, const float* __restrict__ g,
                                const float* __restrict__ b, float* __restrict__ y) {
  int row = blockIdx.x * 256 + threadIdx.x;
  if (row >= NB * NPIX) return;
  const float* xp = x + (size_t)row * 8;
  float s = 0.f, ss = 0.f;
  for (int i = 0; i < 8; i++) { float v = xp[i]; s += v; ss += v * v; }
  float m = s * 0.125f;
  float rstd = rsqrtf(ss * 0.125f - m * m + LN_EPS);
  float* yp = y + (size_t)row * 8;
  for (int i = 0; i < 8; i++) yp[i] = (xp[i] - m) * rstd * g[i] + b[i];
}

// ===========================================================================
// Generic WMMA GEMM: out[row,d] = LN?(X[row,:]) @ W[d,:] + bias[d] (+ ge_bg)
// 64 rows per block, 256 threads = 8 waves, 16 output tiles (2 per wave).
// ===========================================================================
__global__ void gemm64_kernel(const float* __restrict__ X, const float* __restrict__ W,
                              const float* __restrict__ bias,
                              const float* __restrict__ lng, const float* __restrict__ lnb,
                              const float* __restrict__ gew, const float* __restrict__ geb,
                              int useLN, int addGe, float* __restrict__ out) {
  __shared__ alignas(32) _Float16 sA[64][LDSP];
  __shared__ alignas(32) _Float16 sW[64][LDSP];   // sW[d][i] = W[d,i] (row-major f16)
  __shared__ float wx_s[64], wy_s[64], gb_s[64], lg_s[64], lb_s[64], bs_s[64];
  int tid = threadIdx.x;
  int row0 = blockIdx.x * 64;
  if (tid < 64) {
    bs_s[tid] = bias ? bias[tid] : 0.f;
    if (addGe) {
      wx_s[tid] = gew[tid * 4 + 0] - gew[tid * 4 + 2];
      wy_s[tid] = gew[tid * 4 + 1] - gew[tid * 4 + 3];
      gb_s[tid] = geb[tid];
    }
    if (useLN) { lg_s[tid] = lng[tid]; lb_s[tid] = lnb[tid]; }
  }
  // stage W -> f16, 16 contiguous halves per thread (aligned 32B vector store)
  {
    int d = tid >> 2, i0 = (tid & 3) << 4;
    const float* wp = W + d * 64 + i0;
    v16h w16;
    #pragma unroll
    for (int j = 0; j < 16; j++) w16[j] = (_Float16)wp[j];
    *(v16h*)&sW[d][i0] = w16;
  }
  __syncthreads();

  // phase 1: stage (optionally layernormed) input rows as f16
  int r = tid >> 2, seg = (tid & 3) << 4;
  const float* xp = X + (size_t)(row0 + r) * DD;
  __builtin_prefetch(xp + 64 * DD, 0, 3);
  v16h a16;
  if (useLN) {
    float t[16], s = 0.f, ss = 0.f;
    for (int j = 0; j < 16; j++) { float v = xp[seg + j]; t[j] = v; s += v; ss += v * v; }
    s  += __shfl_xor(s, 1);  ss += __shfl_xor(ss, 1);
    s  += __shfl_xor(s, 2);  ss += __shfl_xor(ss, 2);
    float m = s * (1.f / 64.f);
    float rstd = rsqrtf(ss * (1.f / 64.f) - m * m + LN_EPS);
    #pragma unroll
    for (int j = 0; j < 16; j++) {
      int c = seg + j;
      a16[j] = (_Float16)((t[j] - m) * rstd * lg_s[c] + lb_s[c]);
    }
  } else {
    #pragma unroll
    for (int j = 0; j < 16; j++) a16[j] = (_Float16)xp[seg + j];
  }
  *(v16h*)&sA[r][seg] = a16;
  __syncthreads();

  // phase 2: WMMA
  int wave = tid >> 5, lane = tid & 31, lhalf = lane >> 4, lmod = lane & 15;
  for (int tile = wave * 2; tile < wave * 2 + 2; ++tile) {
    int rt = tile >> 2, ct = tile & 3;
    v8f c = {};
    int mm = rt * 16 + lmod;
    int nn = ct * 16 + lmod;
    #pragma unroll
    for (int kc = 0; kc < 2; kc++) {
      v16h a = load_a_frag(&sA[mm][0], kc * 32 + lhalf * 16);
      v16h bb = *(const v16h*)&sW[nn][kc * 32 + lhalf * 16];
      c = __builtin_amdgcn_wmma_f32_16x16x32_f16(false, a, false, bb, (short)0, c, false, false);
    }
    int d = ct * 16 + lmod;
    #pragma unroll
    for (int rr = 0; rr < 8; rr++) {
      int grow = row0 + rt * 16 + lhalf * 8 + rr;
      float v = c[rr] + bs_s[d];
      if (addGe) {
        int n = grow & (NPIX - 1);
        float gx = -1.f + (2.f * (float)(n & 63) + 1.f) * (1.f / 64.f);
        float gy = -1.f + (2.f * (float)(n >> 6) + 1.f) * (1.f / 64.f);
        v += gx * wx_s[d] + gy * wy_s[d] + gb_s[d];
      }
      out[(size_t)grow * DD + d] = v;
    }
  }
}

// ===========================================================================
// q projections: q = LN(slot) @ qW^T  (tiny; one 64-thread block per (b,slot))
// ===========================================================================
__global__ void q_kernel(const float* __restrict__ sfg, const float* __restrict__ sbg,
                         const float* __restrict__ qlng, const float* __restrict__ qlnb,
                         const float* __restrict__ qw,
                         const float* __restrict__ qblng, const float* __restrict__ qblnb,
                         const float* __restrict__ qbw,
                         float* __restrict__ qfg, float* __restrict__ qbg) {
  int blk = blockIdx.x, b = blk >> 3, kk = blk & 7, tid = threadIdx.x;
  __shared__ float row[64], ln[64], mv[2];
  const float *src, *lg, *lb, *W; float* dst;
  if (kk == 0) { src = sbg + b * 64; lg = qblng; lb = qblnb; W = qbw; dst = qbg + b * 64; }
  else { src = sfg + (b * 7 + kk - 1) * 64; lg = qlng; lb = qlnb; W = qw; dst = qfg + (b * 7 + kk - 1) * 64; }
  row[tid] = src[tid];
  __syncthreads();
  if (tid == 0) {
    float s = 0.f, ss = 0.f;
    for (int i = 0; i < 64; i++) { s += row[i]; ss += row[i] * row[i]; }
    float m = s * (1.f / 64.f);
    mv[0] = m; mv[1] = rsqrtf(ss * (1.f / 64.f) - m * m + LN_EPS);
  }
  __syncthreads();
  ln[tid] = (row[tid] - mv[0]) * mv[1] * lg[tid] + lb[tid];
  __syncthreads();
  float acc = 0.f;
  for (int i = 0; i < 64; i++) acc += ln[i] * W[tid * 64 + i];
  dst[tid] = acc;
}

// ===========================================================================
// fused fg logits: per (tile,slot,b): ge -> LN(kbase+ge) -> WMMA*W + b -> q-dot
// ===========================================================================
__global__ void fg_logits_kernel(const float* __restrict__ kbase, const float* __restrict__ W,
                                 const float* __restrict__ bias,
                                 const float* __restrict__ lng, const float* __restrict__ lnb,
                                 const float* __restrict__ gew, const float* __restrict__ geb,
                                 const float* __restrict__ qfg, const float* __restrict__ pos,
                                 float* __restrict__ logits) {
  int tid = threadIdx.x;
  int n0 = blockIdx.x * 64, kslot = blockIdx.y, b = blockIdx.z;
  __shared__ alignas(32) _Float16 sA[64][LDSP];
  __shared__ alignas(32) _Float16 sW[64][LDSP];
  __shared__ float wx_s[64], wy_s[64], gb_s[64], lg_s[64], lb_s[64], mb_s[64], q_s[64];
  __shared__ float lpart[4][64];
  if (tid < 64) {
    wx_s[tid] = gew[tid * 4 + 0] - gew[tid * 4 + 2];
    wy_s[tid] = gew[tid * 4 + 1] - gew[tid * 4 + 3];
    gb_s[tid] = geb[tid];
    lg_s[tid] = lng[tid]; lb_s[tid] = lnb[tid];
    mb_s[tid] = bias[tid];
    q_s[tid] = qfg[(b * 7 + kslot) * 64 + tid];
  }
  {
    int d = tid >> 2, i0 = (tid & 3) << 4;
    const float* wp = W + d * 64 + i0;
    v16h w16;
    #pragma unroll
    for (int j = 0; j < 16; j++) w16[j] = (_Float16)wp[j];
    *(v16h*)&sW[d][i0] = w16;
  }
  __syncthreads();

  float px = pos[(b * 7 + kslot) * 2 + 0];
  float py = pos[(b * 7 + kslot) * 2 + 1];
  int r = tid >> 2, seg = (tid & 3) << 4;
  int n = n0 + r;
  float gx = -1.f + (2.f * (float)(n & 63) + 1.f) * (1.f / 64.f);
  float gy = -1.f + (2.f * (float)(n >> 6) + 1.f) * (1.f / 64.f);
  float rx = gx - px, ry = gy - py;
  const float* bp = kbase + ((size_t)b * NPIX + n) * DD;
  __builtin_prefetch(bp + 64 * DD, 0, 3);
  float t[16], s = 0.f, ss = 0.f;
  for (int j = 0; j < 16; j++) {
    int c = seg + j;
    float v = bp[c] + rx * wx_s[c] + ry * wy_s[c] + gb_s[c];
    t[j] = v; s += v; ss += v * v;
  }
  s  += __shfl_xor(s, 1);  ss += __shfl_xor(ss, 1);
  s  += __shfl_xor(s, 2);  ss += __shfl_xor(ss, 2);
  float m = s * (1.f / 64.f);
  float rstd = rsqrtf(ss * (1.f / 64.f) - m * m + LN_EPS);
  v16h a16;
  #pragma unroll
  for (int j = 0; j < 16; j++) {
    int c = seg + j;
    a16[j] = (_Float16)((t[j] - m) * rstd * lg_s[c] + lb_s[c]);
  }
  *(v16h*)&sA[r][seg] = a16;
  __syncthreads();

  int wave = tid >> 5, lane = tid & 31, lhalf = lane >> 4, lmod = lane & 15;
  for (int tile = wave * 2; tile < wave * 2 + 2; ++tile) {
    int rt = tile >> 2, ct = tile & 3;
    v8f c = {};
    int mm = rt * 16 + lmod;
    int nn = ct * 16 + lmod;
    #pragma unroll
    for (int kc = 0; kc < 2; kc++) {
      v16h a = load_a_frag(&sA[mm][0], kc * 32 + lhalf * 16);
      v16h bb = *(const v16h*)&sW[nn][kc * 32 + lhalf * 16];
      c = __builtin_amdgcn_wmma_f32_16x16x32_f16(false, a, false, bb, (short)0, c, false, false);
    }
    int d = ct * 16 + lmod;
    float qd = q_s[d], bd = mb_s[d];
    #pragma unroll
    for (int rr = 0; rr < 8; rr++) {
      float val = (c[rr] + bd) * qd;
      val += __shfl_xor(val, 1);
      val += __shfl_xor(val, 2);
      val += __shfl_xor(val, 4);
      val += __shfl_xor(val, 8);
      if (lmod == 0) lpart[ct][rt * 16 + lhalf * 8 + rr] = val;
    }
  }
  __syncthreads();
  if (tid < 64)
    logits[((size_t)b * NK + 1 + kslot) * NPIX + n0 + tid] =
        (lpart[0][tid] + lpart[1][tid] + lpart[2][tid] + lpart[3][tid]) * SCALE;
}

// ===========================================================================
// bg logits: dot(q_bg, k_bg) * SCALE
// ===========================================================================
__global__ void bg_logits_kernel(const float* __restrict__ qbg, const float* __restrict__ kbg,
                                 float* __restrict__ logits) {
  int idx = blockIdx.x * 256 + threadIdx.x;
  if (idx >= NB * NPIX) return;
  int b = idx >> 12, n = idx & (NPIX - 1);
  const float* q = qbg + b * 64;
  const float* kk = kbg + (size_t)idx * 64;
  float s = 0.f;
  for (int d = 0; d < 64; d++) s += q[d] * kk[d];
  logits[(size_t)b * NK * NPIX + n] = s * SCALE;
}

// ===========================================================================
// softmax across 8 slots (+EPS), written into the attn output region
// ===========================================================================
__global__ void softmax_kernel(const float* __restrict__ logits, float* __restrict__ attn) {
  int idx = blockIdx.x * 256 + threadIdx.x;
  if (idx >= NB * NPIX) return;
  int b = idx >> 12, n = idx & (NPIX - 1);
  const float* lp = logits + (size_t)b * NK * NPIX + n;
  float l[NK], m = -1e30f;
  for (int k = 0; k < NK; k++) { l[k] = lp[(size_t)k * NPIX]; m = fmaxf(m, l[k]); }
  float s = 0.f;
  for (int k = 0; k < NK; k++) { l[k] = __expf(l[k] - m); s += l[k]; }
  float inv = 1.f / s;
  float* ap = attn + (size_t)b * NK * NPIX + n;
  for (int k = 0; k < NK; k++) ap[(size_t)k * NPIX] = l[k] * inv + EPS_ATTN;
}

// ===========================================================================
// per-(b,slot) sum of attn over N
// ===========================================================================
__global__ void rowsum_kernel(const float* __restrict__ attn, float* __restrict__ sums) {
  int blk = blockIdx.x, tid = threadIdx.x;
  __shared__ float sh[256];
  float s = 0.f;
  const float* ap = attn + (size_t)blk * NPIX;
  for (int n = tid; n < NPIX; n += 256) s += ap[n];
  sh[tid] = s;
  __syncthreads();
  for (int off = 128; off > 0; off >>= 1) {
    if (tid < off) sh[tid] += sh[tid + off];
    __syncthreads();
  }
  if (tid == 0) sums[blk] = sh[0];
}

// ===========================================================================
// fg_position update: sum w*grid + (w @ pos_bias_w^T)/5
// ===========================================================================
__global__ void pos_kernel(const float* __restrict__ attn, const float* __restrict__ sums,
                           const float* __restrict__ pbw,
                           float* __restrict__ pos, float* __restrict__ pos_out) {
  int blk = blockIdx.x, b = blk / 7, k = blk % 7, tid = threadIdx.x;
  __shared__ float sh[4][256];
  float inv = 1.f / sums[b * NK + 1 + k];
  const float* ap = attn + ((size_t)b * NK + 1 + k) * NPIX;
  float a0 = 0.f, a1 = 0.f, a2 = 0.f, a3 = 0.f;
  for (int n = tid; n < NPIX; n += 256) {
    float w = ap[n] * inv;
    float gx = -1.f + (2.f * (float)(n & 63) + 1.f) * (1.f / 64.f);
    float gy = -1.f + (2.f * (float)(n >> 6) + 1.f) * (1.f / 64.f);
    a0 += w * gx; a1 += w * gy; a2 += w * pbw[n]; a3 += w * pbw[NPIX + n];
  }
  sh[0][tid] = a0; sh[1][tid] = a1; sh[2][tid] = a2; sh[3][tid] = a3;
  __syncthreads();
  for (int off = 128; off > 0; off >>= 1) {
    if (tid < off)
      for (int c = 0; c < 4; c++) sh[c][tid] += sh[c][tid + off];
    __syncthreads();
  }
  if (tid == 0) {
    float px = sh[0][0] + sh[2][0] * 0.2f;
    float py = sh[1][0] + sh[3][0] * 0.2f;
    pos[blk * 2 + 0] = px; pos[blk * 2 + 1] = py;
    pos_out[blk * 2 + 0] = px; pos_out[blk * 2 + 1] = py;
  }
}

// ===========================================================================
// fused fg update pass: recompute v tile via WMMA, contract with w_fg,
// write deterministic per-tile partials.
// ===========================================================================
__global__ void fg_upd_kernel(const float* __restrict__ vbase, const float* __restrict__ W,
                              const float* __restrict__ bias,
                              const float* __restrict__ lng, const float* __restrict__ lnb,
                              const float* __restrict__ gew, const float* __restrict__ geb,
                              const float* __restrict__ pos,
                              const float* __restrict__ attn, const float* __restrict__ sums,
                              float* __restrict__ updpart) {
  int tid = threadIdx.x;
  int tileIdx = blockIdx.x, kslot = blockIdx.y, b = blockIdx.z;
  int n0 = tileIdx * 64;
  __shared__ alignas(32) _Float16 sA[64][LDSP];
  __shared__ alignas(32) _Float16 sW[64][LDSP];
  __shared__ float wx_s[64], wy_s[64], gb_s[64], lg_s[64], lb_s[64], mb_s[64], wrow[64];
  __shared__ float upart[4][64];
  if (tid < 64) {
    wx_s[tid] = gew[tid * 4 + 0] - gew[tid * 4 + 2];
    wy_s[tid] = gew[tid * 4 + 1] - gew[tid * 4 + 3];
    gb_s[tid] = geb[tid];
    lg_s[tid] = lng[tid]; lb_s[tid] = lnb[tid];
    mb_s[tid] = bias[tid];
    wrow[tid] = attn[((size_t)b * NK + 1 + kslot) * NPIX + n0 + tid] /
                sums[b * NK + 1 + kslot];
  }
  {
    int d = tid >> 2, i0 = (tid & 3) << 4;
    const float* wp = W + d * 64 + i0;
    v16h w16;
    #pragma unroll
    for (int j = 0; j < 16; j++) w16[j] = (_Float16)wp[j];
    *(v16h*)&sW[d][i0] = w16;
  }
  __syncthreads();

  float px = pos[(b * 7 + kslot) * 2 + 0];
  float py = pos[(b * 7 + kslot) * 2 + 1];
  int r = tid >> 2, seg = (tid & 3) << 4;
  int n = n0 + r;
  float gx = -1.f + (2.f * (float)(n & 63) + 1.f) * (1.f / 64.f);
  float gy = -1.f + (2.f * (float)(n >> 6) + 1.f) * (1.f / 64.f);
  float rx = gx - px, ry = gy - py;
  const float* bp = vbase + ((size_t)b * NPIX + n) * DD;
  __builtin_prefetch(bp + 64 * DD, 0, 3);
  float t[16], s = 0.f, ss = 0.f;
  for (int j = 0; j < 16; j++) {
    int c = seg + j;
    float v = bp[c] + rx * wx_s[c] + ry * wy_s[c] + gb_s[c];
    t[j] = v; s += v; ss += v * v;
  }
  s  += __shfl_xor(s, 1);  ss += __shfl_xor(ss, 1);
  s  += __shfl_xor(s, 2);  ss += __shfl_xor(ss, 2);
  float m = s * (1.f / 64.f);
  float rstd = rsqrtf(ss * (1.f / 64.f) - m * m + LN_EPS);
  v16h a16;
  #pragma unroll
  for (int j = 0; j < 16; j++) {
    int c = seg + j;
    a16[j] = (_Float16)((t[j] - m) * rstd * lg_s[c] + lb_s[c]);
  }
  *(v16h*)&sA[r][seg] = a16;
  __syncthreads();

  int wave = tid >> 5, lane = tid & 31, lhalf = lane >> 4, lmod = lane & 15;
  for (int tile = wave * 2; tile < wave * 2 + 2; ++tile) {
    int rt = tile >> 2, ct = tile & 3;
    v8f c = {};
    int mm = rt * 16 + lmod;
    int nn = ct * 16 + lmod;
    #pragma unroll
    for (int kc = 0; kc < 2; kc++) {
      v16h a = load_a_frag(&sA[mm][0], kc * 32 + lhalf * 16);
      v16h bb = *(const v16h*)&sW[nn][kc * 32 + lhalf * 16];
      c = __builtin_amdgcn_wmma_f32_16x16x32_f16(false, a, false, bb, (short)0, c, false, false);
    }
    int d = ct * 16 + lmod;
    float bd = mb_s[d], part = 0.f;
    #pragma unroll
    for (int rr = 0; rr < 8; rr++)
      part += (c[rr] + bd) * wrow[rt * 16 + lhalf * 8 + rr];
    part += __shfl_xor(part, 16);
    if (lhalf == 0) upart[rt][d] = part;
  }
  __syncthreads();
  if (tid < 64)
    updpart[(((size_t)b * 7 + kslot) * 64 + tileIdx) * 64 + tid] =
        upart[0][tid] + upart[1][tid] + upart[2][tid] + upart[3][tid];
}

__global__ void upd_reduce_kernel(const float* __restrict__ updpart, float* __restrict__ ufg) {
  int blk = blockIdx.x, tid = threadIdx.x;  // 56 blocks, 64 threads
  float s = 0.f;
  for (int t = 0; t < 64; t++) s += updpart[((size_t)blk * 64 + t) * 64 + tid];
  ufg[blk * 64 + tid] = s;
}

// ===========================================================================
// bg update: upd_bg[b,d] = sum_n w_bg * v_bg
// ===========================================================================
__global__ void bg_upd_kernel(const float* __restrict__ attn, const float* __restrict__ sums,
                              const float* __restrict__ vbg, float* __restrict__ ubg) {
  int b = blockIdx.x, tid = threadIdx.x;
  int d = tid & 63, g = tid >> 6;
  __shared__ float sh[256];
  float inv = 1.f / sums[b * NK];
  const float* ap = attn + (size_t)b * NK * NPIX;
  float s = 0.f;
  for (int n = g; n < NPIX; n += 4)
    s += ap[n] * inv * vbg[((size_t)b * NPIX + n) * DD + d];
  sh[tid] = s;
  __syncthreads();
  if (g == 0) ubg[b * 64 + d] = sh[d] + sh[64 + d] + sh[128 + d] + sh[192 + d];
}

// ===========================================================================
// GRU cell + residual MLP, per (b,slot); slot 0 = background params
// ===========================================================================
__global__ void gru_kernel(const float* __restrict__ ufg, const float* __restrict__ ubg,
                           const float* __restrict__ wihf, const float* __restrict__ whhf,
                           const float* __restrict__ bihf, const float* __restrict__ bhhf,
                           const float* __restrict__ wihb, const float* __restrict__ whhb,
                           const float* __restrict__ bihb, const float* __restrict__ bhhb,
                           const float* __restrict__ rflng, const float* __restrict__ rflnb,
                           const float* __restrict__ rfw, const float* __restrict__ rfb,
                           const float* __restrict__ rblng, const float* __restrict__ rblnb,
                           const float* __restrict__ rbw, const float* __restrict__ rbb,
                           float* __restrict__ sfg, float* __restrict__ sbg) {
  int blk = blockIdx.x, b = blk >> 3, kk = blk & 7, tid = threadIdx.x;  // 192 threads
  __shared__ float x_s[64], h_s[64], gi_s[192], gh_s[192], hn_s[64], ln_s[64], mv[2];
  const float *wih, *whh, *bih, *bhh, *rlg, *rlb, *rw, *rb, *updp;
  float* slotp;
  if (kk == 0) {
    wih = wihb; whh = whhb; bih = bihb; bhh = bhhb;
    rlg = rblng; rlb = rblnb; rw = rbw; rb = rbb;
    slotp = sbg + b * 64; updp = ubg + b * 64;
  } else {
    wih = wihf; whh = whhf; bih = bihf; bhh = bhhf;
    rlg = rflng; rlb = rflnb; rw = rfw; rb = rfb;
    slotp = sfg + (b * 7 + kk - 1) * 64; updp = ufg + (b * 7 + kk - 1) * 64;
  }
  if (tid < 64) { x_s[tid] = updp[tid]; h_s[tid] = slotp[tid]; }
  __syncthreads();
  float gi = bih[tid], gh = bhh[tid];
  for (int i = 0; i < 64; i++) {
    gi += x_s[i] * wih[tid * 64 + i];
    gh += h_s[i] * whh[tid * 64 + i];
  }
  gi_s[tid] = gi; gh_s[tid] = gh;
  __syncthreads();
  if (tid < 64) {
    float rr = 1.f / (1.f + __expf(-(gi_s[tid] + gh_s[tid])));
    float zz = 1.f / (1.f + __expf(-(gi_s[64 + tid] + gh_s[64 + tid])));
    float nn = tanhf(gi_s[128 + tid] + rr * gh_s[128 + tid]);
    hn_s[tid] = (1.f - zz) * nn + zz * h_s[tid];
  }
  __syncthreads();
  if (tid == 0) {
    float s = 0.f, ss = 0.f;
    for (int i = 0; i < 64; i++) { s += hn_s[i]; ss += hn_s[i] * hn_s[i]; }
    float m = s * (1.f / 64.f);
    mv[0] = m; mv[1] = rsqrtf(ss * (1.f / 64.f) - m * m + LN_EPS);
  }
  __syncthreads();
  if (tid < 64) ln_s[tid] = (hn_s[tid] - mv[0]) * mv[1] * rlg[tid] + rlb[tid];
  __syncthreads();
  if (tid < 64) {
    float acc = rb[tid];
    for (int i = 0; i < 64; i++) acc += ln_s[i] * rw[tid * 64 + i];
    slotp[tid] = hn_s[tid] + acc;
  }
}

// ===========================================================================
// final: slot colors (w-weighted mean of normalized colors) + slot assembly
// ===========================================================================
__global__ void color_kernel(const float* __restrict__ attn, const float* __restrict__ sums,
                             const float* __restrict__ colorn, float* __restrict__ out_slots) {
  int blk = blockIdx.x, b = blk >> 3, tid = threadIdx.x;
  __shared__ float sh[256][8];
  float acc[8] = {0.f, 0.f, 0.f, 0.f, 0.f, 0.f, 0.f, 0.f};
  float inv = 1.f / sums[blk];
  const float* ap = attn + (size_t)blk * NPIX;
  for (int n = tid; n < NPIX; n += 256) {
    float w = ap[n] * inv;
    const float* cp = colorn + ((size_t)b * NPIX + n) * 8;
    for (int c = 0; c < 8; c++) acc[c] += w * cp[c];
  }
  for (int c = 0; c < 8; c++) sh[tid][c] = acc[c];
  __syncthreads();
  for (int off = 128; off > 0; off >>= 1) {
    if (tid < off)
      for (int c = 0; c < 8; c++) sh[tid][c] += sh[tid + off][c];
    __syncthreads();
  }
  if (tid < 8) out_slots[(size_t)blk * 72 + 64 + tid] = sh[0][tid];
}

__global__ void slots_write_kernel(const float* __restrict__ sfg, const float* __restrict__ sbg,
                                   float* __restrict__ out_slots) {
  int blk = blockIdx.x, b = blk >> 3, kk = blk & 7, tid = threadIdx.x;  // 64 threads
  float v = (kk == 0) ? sbg[b * 64 + tid] : sfg[(b * 7 + kk - 1) * 64 + tid];
  out_slots[(size_t)blk * 72 + tid] = v;
}

// ===========================================================================
extern "C" void kernel_launch(void* const* d_in, const int* in_sizes, int n_in,
                              void* d_out, int out_size, void* d_ws, size_t ws_size,
                              hipStream_t stream) {
  (void)in_sizes; (void)n_in; (void)out_size; (void)ws_size;
  #define IN(i) ((const float*)d_in[i])
  const float *feat = IN(0), *featc = IN(1), *noisefg = IN(2), *noisebg = IN(3);
  const float *mu = IN(4), *ls = IN(5), *mubg = IN(6), *lsbg = IN(7);
  const float *pos0 = IN(8), *pbw = IN(9), *gew = IN(10), *geb = IN(11);
  const float *kfgw = IN(12), *vfgw = IN(13), *kbgw = IN(14), *vbgw = IN(15);
  const float *fglng = IN(16), *fglnb = IN(17), *fgw = IN(18), *fgb = IN(19);
  const float *bglng = IN(20), *bglnb = IN(21), *bgw = IN(22), *bgb = IN(23);
  const float *qlng = IN(24), *qlnb = IN(25), *qw = IN(26);
  const float *qblng = IN(27), *qblnb = IN(28), *qbw = IN(29);
  const float *wihf = IN(30), *whhf = IN(31), *bihf = IN(32), *bhhf = IN(33);
  const float *wihb = IN(34), *whhb = IN(35), *bihb = IN(36), *bhhb = IN(37);
  const float *rflng = IN(38), *rflnb = IN(39), *rfw = IN(40), *rfb = IN(41);
  const float *rblng = IN(42), *rblnb = IN(43), *rbw = IN(44), *rbb = IN(45);
  const float *ftlng = IN(46), *ftlnb = IN(47), *clng = IN(48), *clnb = IN(49);
  #undef IN

  float* ws = (float*)d_ws;
  float* featn  = ws + OF_FEATN;
  float* colorn = ws + OF_COLORN;
  float* kfgb   = ws + OF_KFGB;
  float* vfgb   = ws + OF_VFGB;
  float* kbt    = ws + OF_KBT;
  float* vbt    = ws + OF_VBT;
  float* kbg    = ws + OF_KBG;
  float* vbg    = ws + OF_VBG;
  float* logits = ws + OF_LOGITS;
  float* sums   = ws + OF_SUMS;
  float* posA   = ws + OF_POSA;
  float* posB   = ws + OF_POSB;
  float* qfg    = ws + OF_QFG;
  float* qbg    = ws + OF_QBG;
  float* sfg    = ws + OF_SFG;
  float* sbg    = ws + OF_SBG;
  float* ufg    = ws + OF_UFG;
  float* ubg    = ws + OF_UBG;
  float* upart  = ws + OF_UPART;

  float* out      = (float*)d_out;
  float* attn_out = out + (size_t)NB * NK * 72;                 // 4608
  float* pos_out  = attn_out + (size_t)NB * NK * NPIX;          // 266752

  // ---- prep ----
  init_kernel<<<14, 256, 0, stream>>>(mu, ls, mubg, lsbg, noisefg, noisebg, pos0,
                                      sfg, sbg, posA);
  ln_feat_kernel<<<128, 256, 0, stream>>>(feat, ftlng, ftlnb, featn);
  ln_color_kernel<<<128, 256, 0, stream>>>(featc, clng, clnb, colorn);
  gemm64_kernel<<<512, 256, 0, stream>>>(featn, kfgw, nullptr, nullptr, nullptr,
                                         gew, geb, 0, 0, kfgb);
  gemm64_kernel<<<512, 256, 0, stream>>>(featn, vfgw, nullptr, nullptr, nullptr,
                                         gew, geb, 0, 0, vfgb);
  gemm64_kernel<<<512, 256, 0, stream>>>(featn, kbgw, nullptr, nullptr, nullptr,
                                         gew, geb, 0, 1, kbt);
  gemm64_kernel<<<512, 256, 0, stream>>>(featn, vbgw, nullptr, nullptr, nullptr,
                                         gew, geb, 0, 1, vbt);
  gemm64_kernel<<<512, 256, 0, stream>>>(kbt, bgw, bgb, bglng, bglnb,
                                         gew, geb, 1, 0, kbg);
  gemm64_kernel<<<512, 256, 0, stream>>>(vbt, bgw, bgb, bglng, bglnb,
                                         gew, geb, 1, 0, vbg);

  float* pcur = posA;
  float* pnext = posB;
  for (int it = 0; it < 4; ++it) {
    q_kernel<<<NB * NK, 64, 0, stream>>>(sfg, sbg, qlng, qlnb, qw, qblng, qblnb, qbw,
                                         qfg, qbg);
    fg_logits_kernel<<<dim3(64, 7, NB), 256, 0, stream>>>(
        kfgb, fgw, fgb, fglng, fglnb, gew, geb, qfg, pcur, logits);
    bg_logits_kernel<<<128, 256, 0, stream>>>(qbg, kbg, logits);
    softmax_kernel<<<128, 256, 0, stream>>>(logits, attn_out);
    rowsum_kernel<<<NB * NK, 256, 0, stream>>>(attn_out, sums);
    pos_kernel<<<NB * 7, 256, 0, stream>>>(attn_out, sums, pbw, pnext, pos_out);
    if (it < 3) {
      fg_upd_kernel<<<dim3(64, 7, NB), 256, 0, stream>>>(
          vfgb, fgw, fgb, fglng, fglnb, gew, geb, pcur, attn_out, sums, upart);
      upd_reduce_kernel<<<NB * 7, 64, 0, stream>>>(upart, ufg);
      bg_upd_kernel<<<NB, 256, 0, stream>>>(attn_out, sums, vbg, ubg);
      gru_kernel<<<NB * NK, 192, 0, stream>>>(ufg, ubg,
                                              wihf, whhf, bihf, bhhf,
                                              wihb, whhb, bihb, bhhb,
                                              rflng, rflnb, rfw, rfb,
                                              rblng, rblnb, rbw, rbb,
                                              sfg, sbg);
    } else {
      color_kernel<<<NB * NK, 256, 0, stream>>>(attn_out, sums, colorn, out);
      slots_write_kernel<<<NB * NK, 64, 0, stream>>>(sfg, sbg, out);
    }
    float* tmp = pcur; pcur = pnext; pnext = tmp;
  }
}
